// ESN_35983236006615
// MI455X (gfx1250) — matrix operand order repaired and checked by reading, not compiled, not run on hardware
//
#include <hip/hip_runtime.h>
#include <hip/hip_bf16.h>
#include <math.h>

// ---------------------------------------------------------------------------
// ESN teacher-forced rollout for MI455X (gfx1250).
//   Phase 1: convert W_res (256 MB fp32, NT reads) -> bf16 (128 MB, L2-resident).
//   Phase 2: in_proj = in_seq @ W_in^T + b  via v_wmma_f32_16x16x32_bf16
//            (regular-temporal stores: 32 MB kept in L2 alongside 128 MB W).
//   Phase 3: 1024 sequential steps r' = tanh(W_res@r + u_t); bf16 WMMA matvec,
//            8-way K-split per 16-output block, LDS reduction. in_proj rows
//            are last-use NT loads; fp32 outputs are NT stores so the W matrix
//            is never evicted from L2.
// ---------------------------------------------------------------------------

#define SEQ_LEN   1024
#define STATE_DIM 128
#define RES_DIM   8192

typedef __attribute__((ext_vector_type(16))) __bf16 v16bf;
typedef __attribute__((ext_vector_type(8)))  __bf16 v8bf;
typedef __attribute__((ext_vector_type(8)))  float  v8f;
typedef __attribute__((ext_vector_type(4)))  float  v4f;

// ---------------------------------------------------------------------------
// Prep: W_res fp32 -> bf16 (8 elems/thread); also res_state fp32 -> bf16 r0.
// fp32 source is streamed with NT loads (read-once, 256 MB).
// ---------------------------------------------------------------------------
__global__ void esn_prep(const float* __restrict__ Wres,
                         const float* __restrict__ r0,
                         __bf16* __restrict__ W16,
                         __bf16* __restrict__ rbuf) {
    size_t idx  = (size_t)blockIdx.x * blockDim.x + threadIdx.x;
    size_t base = idx * 8;
    const size_t NW = (size_t)RES_DIM * RES_DIM;
    if (base < NW) {
        v4f a = __builtin_nontemporal_load((const v4f*)(Wres + base));
        v4f b = __builtin_nontemporal_load((const v4f*)(Wres + base + 4));
        v8bf o;
        o[0] = (__bf16)a[0]; o[1] = (__bf16)a[1];
        o[2] = (__bf16)a[2]; o[3] = (__bf16)a[3];
        o[4] = (__bf16)b[0]; o[5] = (__bf16)b[1];
        o[6] = (__bf16)b[2]; o[7] = (__bf16)b[3];
        *(v8bf*)(W16 + base) = o;      // regular: keep bf16 W in L2
    }
    if (idx < RES_DIM / 8) {
        size_t rb = idx * 8;
        v4f a = *(const v4f*)(r0 + rb);
        v4f b = *(const v4f*)(r0 + rb + 4);
        v8bf o;
        o[0] = (__bf16)a[0]; o[1] = (__bf16)a[1];
        o[2] = (__bf16)a[2]; o[3] = (__bf16)a[3];
        o[4] = (__bf16)b[0]; o[5] = (__bf16)b[1];
        o[6] = (__bf16)b[2]; o[7] = (__bf16)b[3];
        *(v8bf*)(rbuf + rb) = o;
    }
}

// ---------------------------------------------------------------------------
// in_proj[t][i] = sum_s in_seq[t][s] * W_in[i][s] + b_res[i]
// One 16x16 output tile per wave; K=128 -> 4 WMMA steps of K=32.
// A layout (16x32 bf16, MxK): lane L (m=L&15, h=L>>4) holds K = 16h+j
//   (j=0..15) as vector elems [0..15].
// B layout (32x16 bf16, KxN): lane L (n=L&15, h=L>>4) holds K = 16h+j
//   (j=0..15) of column n.
// ---------------------------------------------------------------------------
__global__ void esn_inproj(const float* __restrict__ in_seq,
                           const float* __restrict__ W_in,
                           const float* __restrict__ b_res,
                           float* __restrict__ in_proj) {
    const int lane = threadIdx.x & 31;
    const int wave = threadIdx.x >> 5;
    const int tb   = blockIdx.x >> 6;              // 64 t-tiles
    const int ig   = blockIdx.x & 63;              // 64 i-groups of 8 tiles
    const int t0   = tb * 16;
    const int i0   = (ig * 8 + wave) * 16;
    const int m    = lane & 15;
    const int h    = lane >> 4;

    v8f c = {};
#pragma unroll
    for (int kk = 0; kk < STATE_DIM; kk += 32) {
        // A tile: rows of in_seq (fp32 -> bf16 in-register)
        const float* ap0 = in_seq + (size_t)(t0 + m) * STATE_DIM + kk + 16 * h;
        const float* ap1 = ap0 + 8;
        v16bf a;
#pragma unroll
        for (int j = 0; j < 8; ++j) {
            a[j]     = (__bf16)ap0[j];
            a[8 + j] = (__bf16)ap1[j];
        }
        // B tile: B[k][n] = W_in[i0+n][kk+k]
        const float* bp = W_in + (size_t)(i0 + m) * STATE_DIM + kk + 16 * h;
        v16bf b;
#pragma unroll
        for (int j = 0; j < 16; ++j) b[j] = (__bf16)bp[j];

        c = __builtin_amdgcn_wmma_f32_16x16x32_bf16(
                false, a, false, b, (short)0, c, false, false);
    }
    // D[M][n]: VGPR v holds M = v + 8h, n = lane&15. Regular stores: the
    // 32 MB in_proj tensor stays L2-resident next to the 128 MB bf16 W.
    const float bias = b_res[i0 + m];
#pragma unroll
    for (int v = 0; v < 8; ++v) {
        in_proj[(size_t)(t0 + v + 8 * h) * RES_DIM + i0 + m] = c[v] + bias;
    }
}

// ---------------------------------------------------------------------------
// One scan step: r_out = tanh(W16 @ r_in + uproj); also writes fp32 out row.
// Block = 8 waves = one 16-output block; wave w reduces K in [w*1024,(w+1)*1024).
// B operand = r chunk broadcast to all 16 N-columns (D columns identical).
// ---------------------------------------------------------------------------
__global__ void esn_step(const __bf16* __restrict__ W16,
                         const __bf16* __restrict__ r_in,
                         const float*  __restrict__ uproj,
                         __bf16* __restrict__ r_out,
                         float*  __restrict__ out_row) {
    __shared__ float part[8 * 16];

    const int lane = threadIdx.x & 31;
    const int wave = threadIdx.x >> 5;
    const int i0   = blockIdx.x * 16;
    const int m    = lane & 15;
    const int h    = lane >> 4;

    const __bf16* wrow = W16 + (size_t)(i0 + m) * RES_DIM;
    const int k0 = wave * (RES_DIM / 8);

    v8f c = {};
#pragma unroll 4
    for (int kk = k0; kk < k0 + RES_DIM / 8; kk += 32) {
        // A: W block rows (per-lane two contiguous 16B chunks, L2-resident)
        v8bf a0 = *(const v8bf*)(wrow + kk + 8 * h);
        v8bf a1 = *(const v8bf*)(wrow + kk + 16 + 8 * h);
        v16bf a = __builtin_shufflevector(a0, a1,
                    0,1,2,3,4,5,6,7,8,9,10,11,12,13,14,15);
        // B: r chunk, K = kk + 8h + j and kk + 16 + 8h + j, broadcast across columns
        v8bf b0 = *(const v8bf*)(r_in + kk + 8 * h);
        v8bf b1 = *(const v8bf*)(r_in + kk + 16 + 8 * h);
        v16bf b = __builtin_shufflevector(b0, b1,
                    0,1,2,3,4,5,6,7,8,9,10,11,12,13,14,15);

        c = __builtin_amdgcn_wmma_f32_16x16x32_bf16(
                false, a, false, b, (short)0, c, false, false);
    }

    // D columns are identical; lanes n==0 (L=0,16) hold partials for
    // M = v + 8h. Stash per-wave partials in LDS.
    if (m == 0) {
#pragma unroll
        for (int v = 0; v < 8; ++v) part[wave * 16 + v + 8 * h] = c[v];
    }
    __syncthreads();

    const int tid = threadIdx.x;
    if (tid < 16) {
        // in_proj row: read exactly once ever -> NT (last-use) load.
        float acc = __builtin_nontemporal_load(uproj + i0 + tid);
#pragma unroll
        for (int w = 0; w < 8; ++w) acc += part[w * 16 + tid];
        float y = tanhf(acc);
        // Result row never re-read on device -> NT store (don't evict W).
        __builtin_nontemporal_store(y, out_row + i0 + tid);
        r_out[i0 + tid] = (__bf16)y;   // tiny, re-read next step: regular
    }
}

// ---------------------------------------------------------------------------
extern "C" void kernel_launch(void* const* d_in, const int* in_sizes, int n_in,
                              void* d_out, int out_size, void* d_ws, size_t ws_size,
                              hipStream_t stream) {
    const float* in_seq    = (const float*)d_in[0];   // (1024, 128)
    const float* res_state = (const float*)d_in[1];   // (8192,)
    const float* W_in      = (const float*)d_in[2];   // (8192, 128)
    const float* W_res     = (const float*)d_in[3];   // (8192, 8192)
    const float* b_res     = (const float*)d_in[4];   // (8192,)
    float* out             = (float*)d_out;           // (1024, 8192)

    char* ws = (char*)d_ws;
    __bf16* W16     = (__bf16*)ws;                                   // 128 MB
    float*  in_proj = (float*)(ws + (size_t)RES_DIM * RES_DIM * 2);  // 32 MB
    __bf16* rbuf    = (__bf16*)(ws + (size_t)RES_DIM * RES_DIM * 2
                                   + (size_t)SEQ_LEN * RES_DIM * 4); // 32 KB

    // Phase 1: bf16 conversion (W_res + r0)
    const size_t nconv = (size_t)RES_DIM * RES_DIM / 8;
    esn_prep<<<(unsigned)((nconv + 255) / 256), 256, 0, stream>>>(
        W_res, res_state, W16, rbuf);

    // Phase 2: input projection GEMM (64 t-tiles x 64 i-groups, 8 waves/block)
    esn_inproj<<<64 * 64, 256, 0, stream>>>(in_seq, W_in, b_res, in_proj);

    // Phase 3: sequential scan, ping-pong bf16 state
    for (int t = 0; t < SEQ_LEN; ++t) {
        const __bf16* rin  = rbuf + (size_t)(t & 1) * RES_DIM;
        __bf16*       rout = rbuf + (size_t)((t + 1) & 1) * RES_DIM;
        esn_step<<<RES_DIM / 16, 256, 0, stream>>>(
            W16, rin, in_proj + (size_t)t * RES_DIM, rout,
            out + (size_t)t * RES_DIM);
    }
}
